// MultiHeadAttention_32959579030343
// MI455X (gfx1250) — compile-verified, matrix operand order checked by reading, not run on hardware
//
#include <hip/hip_runtime.h>
#include <hip/hip_bf16.h>
#include <math.h>

typedef __attribute__((ext_vector_type(16))) __bf16 v16bf;
typedef __attribute__((ext_vector_type(8)))  float  v8f;
typedef int v4i_b __attribute__((vector_size(16)));

#define B_DIM  2
#define S_DIM  2048
#define D_DIM  512
#define H_DIM  8
#define HD_DIM 64

// ---------------------------------------------------------------------------
// gfx1250 async global->LDS copy (ASYNCcnt-tracked). ROCm 7.2 signature takes
// (v4i addrspace(1)*, v4i addrspace(3)*, imm offset, imm cpol). Fall back to a
// synchronous staging copy if the builtin is unavailable.
// ---------------------------------------------------------------------------
#if defined(__has_builtin)
#if __has_builtin(__builtin_amdgcn_global_load_async_to_lds_b128)
#define ATHENA_ASYNC_LDS 1
#endif
#if __has_builtin(__builtin_amdgcn_s_wait_asynccnt)
#define ATHENA_WAIT_BUILTIN 1
#endif
#endif

static __device__ inline void async_copy_b128(const __bf16* g, __bf16* l) {
#if defined(ATHENA_ASYNC_LDS)
    __builtin_amdgcn_global_load_async_to_lds_b128(
        (__attribute__((address_space(1))) v4i_b*)(g),
        (__attribute__((address_space(3))) v4i_b*)(l), 0, 0);
#else
    *(uint4*)l = *(const uint4*)g;   // synchronous fallback
#endif
}

static __device__ inline void wait_async0() {
#if defined(ATHENA_WAIT_BUILTIN)
    __builtin_amdgcn_s_wait_asynccnt(0);
#else
    asm volatile("s_wait_asynccnt 0" ::: "memory");
#endif
}

// ---------------------------------------------------------------------------
// WMMA fragment loaders for v_wmma_f32_16x16x32_bf16 (wave32 layouts, ISA 7.12.2)
// A (16x32, MxK): lane<16 -> row=lane, K=0..7 & 16..23 ; lane>=16 -> row=lane-16, K=8..15 & 24..31
// B (32x16, KxN): lane<16 -> col=lane, same K packing.
// Loading a row-major Nx32 "B-transposed" source uses exactly the A pattern.
// ---------------------------------------------------------------------------
static __device__ inline v16bf load_a_f32(const float* __restrict__ src, int ld, int lane) {
    const int row = lane & 15;
    const int kb  = (lane >> 4) << 3;
    v16bf a;
#pragma unroll
    for (int i = 0; i < 8; ++i) {
        a[i]     = (__bf16)src[row * ld + kb + i];
        a[i + 8] = (__bf16)src[row * ld + kb + 16 + i];
    }
    return a;
}

static __device__ inline v16bf load_b_f32(const float* __restrict__ src, int ld, int lane) {
    const int col = lane & 15;
    const int kb  = (lane >> 4) << 3;
    v16bf b;
#pragma unroll
    for (int i = 0; i < 8; ++i) {
        b[i]     = (__bf16)src[(kb + i) * ld + col];
        b[i + 8] = (__bf16)src[(kb + 16 + i) * ld + col];
    }
    return b;
}

static __device__ inline v16bf load_a_bf16(const __bf16* src, int ld, int lane) {
    const int row = lane & 15;
    const int kb  = (lane >> 4) << 3;
    v16bf a;
#pragma unroll
    for (int i = 0; i < 8; ++i) {
        a[i]     = src[row * ld + kb + i];
        a[i + 8] = src[row * ld + kb + 16 + i];
    }
    return a;
}

static __device__ inline v16bf load_b_bf16(const __bf16* src, int ld, int lane) {
    const int col = lane & 15;
    const int kb  = (lane >> 4) << 3;
    v16bf b;
#pragma unroll
    for (int i = 0; i < 8; ++i) {
        b[i]     = src[(kb + i) * ld + col];
        b[i + 8] = src[(kb + 16 + i) * ld + col];
    }
    return b;
}

// ---------------------------------------------------------------------------
// Kernel 1: fused Q/K/V projection GEMMs (X[4096x512] @ W[512x512]) in bf16 WMMA,
// output stored bf16 as [proj][B][H][S][HD]. One wave per 16x64 output tile.
// ---------------------------------------------------------------------------
__global__ __launch_bounds__(32) void mha_proj_kernel(
    const float* __restrict__ q, const float* __restrict__ k, const float* __restrict__ v,
    const float* __restrict__ wq, const float* __restrict__ wk, const float* __restrict__ wv,
    __bf16* __restrict__ qkv)
{
    const int lane = threadIdx.x & 31;
    const int m0 = blockIdx.x * 16;       // output row tile (over B*S = 4096)
    const int n0 = blockIdx.y * 64;       // output col group (over 512)
    const int z  = blockIdx.z;            // 0=Q, 1=K, 2=V

    const float* X = (z == 0) ? q  : (z == 1) ? k  : v;
    const float* W = (z == 0) ? wq : (z == 1) ? wk : wv;

    v8f zero = {};
    v8f acc[4] = {zero, zero, zero, zero};

    for (int k0 = 0; k0 < D_DIM; k0 += 32) {
        v16bf a = load_a_f32(X + (size_t)m0 * D_DIM + k0, D_DIM, lane);
#pragma unroll
        for (int t = 0; t < 4; ++t) {
            v16bf b = load_b_f32(W + (size_t)k0 * D_DIM + n0 + t * 16, D_DIM, lane);
            acc[t] = __builtin_amdgcn_wmma_f32_16x16x32_bf16(
                false, a, false, b, (short)0, acc[t], false, false);
        }
    }

    const int col = lane & 15;
    const int rb  = (lane >> 4) << 3;
#pragma unroll
    for (int t = 0; t < 4; ++t) {
#pragma unroll
        for (int r = 0; r < 8; ++r) {
            const int row = m0 + rb + r;              // 0..4095
            const int c   = n0 + t * 16 + col;        // 0..511
            const int bb  = row >> 11;                // /S_DIM
            const int s   = row & (S_DIM - 1);
            const int h   = c >> 6;                   // /HD_DIM
            const int d   = c & (HD_DIM - 1);
            const size_t off =
                ((((size_t)z * B_DIM + bb) * H_DIM + h) * S_DIM + s) * HD_DIM + d;
            qkv[off] = (__bf16)acc[t][r];
        }
    }
}

// ---------------------------------------------------------------------------
// Kernel 2: flash-attention with online softmax.
// Block = 8 waves, all sharing one (b,h); each wave owns a 16-q-row tile.
// K/V streamed through double-buffered LDS via GLOBAL_LOAD_ASYNC_TO_LDS_B128:
// chunk i+1 copies while chunk i feeds WMMAs; one s_wait_asynccnt+barrier/iter.
// ---------------------------------------------------------------------------
__global__ __launch_bounds__(256) void mha_attn_kernel(
    const __bf16* __restrict__ qkv, float* __restrict__ out)
{
    __shared__ __bf16 kbuf[2][32 * HD_DIM];   // 2 x 4 KB
    __shared__ __bf16 vbuf[2][32 * HD_DIM];   // 2 x 4 KB
    __shared__ __bf16 ptile[8][16 * 32];      // per-wave P transpose, 8 KB

    const int tid  = threadIdx.x;
    const int lane = tid & 31;
    const int wave = tid >> 5;
    const int tile = blockIdx.x * 8 + wave;           // 0 .. B*H*(S/16)-1
    const int qt   = tile & (S_DIM / 16 - 1);         // q tile within sequence
    const int bh   = tile >> 7;                       // b*H + h (uniform per block)

    const size_t head_sz = (size_t)S_DIM * HD_DIM;
    const __bf16* Q = qkv + ((size_t)0 * (B_DIM * H_DIM) + bh) * head_sz;
    const __bf16* K = qkv + ((size_t)1 * (B_DIM * H_DIM) + bh) * head_sz;
    const __bf16* V = qkv + ((size_t)2 * (B_DIM * H_DIM) + bh) * head_sz;

    const int q0 = qt * 16;
    const v16bf aq0 = load_a_bf16(Q + (size_t)q0 * HD_DIM + 0,  HD_DIM, lane);
    const v16bf aq1 = load_a_bf16(Q + (size_t)q0 * HD_DIM + 32, HD_DIM, lane);

    v8f zero = {};
    v8f acc[4] = {zero, zero, zero, zero};
    v8f m, l;
#pragma unroll
    for (int r = 0; r < 8; ++r) { m[r] = -INFINITY; l[r] = 0.0f; }

    __bf16* tl = ptile[wave];
    const int col = lane & 15;
    const int rb  = (lane >> 4) << 3;
    const float scale = 0.125f;  // 1/sqrt(HD)

    // Prologue: each thread DMAs one 16B slice of K-chunk 0 and V-chunk 0.
    // 256 threads x 8 bf16 covers the full 32x64 chunk exactly.
    async_copy_b128(K + (size_t)tid * 8, kbuf[0] + tid * 8);
    async_copy_b128(V + (size_t)tid * 8, vbuf[0] + tid * 8);

    int buf = 0;
    for (int kv = 0; kv < S_DIM; kv += 32) {
        wait_async0();        // my slices of chunk kv have landed in LDS
        __syncthreads();      // everyone's slices landed; everyone done with buf^1

        if (kv + 32 < S_DIM) {  // kick off chunk kv+32 into the other buffer
            const size_t goff = (size_t)(kv + 32) * HD_DIM + (size_t)tid * 8;
            async_copy_b128(K + goff, kbuf[buf ^ 1] + tid * 8);
            async_copy_b128(V + goff, vbuf[buf ^ 1] + tid * 8);
        }

        const __bf16* kc = kbuf[buf];
        const __bf16* vc = vbuf[buf];

        // ---- scores: S = Q K^T for keys [kv, kv+32) (K tiles from LDS) ----
        v8f s0 = zero, s1 = zero;
        {
            v16bf bk = load_a_bf16(kc + 0, HD_DIM, lane);
            s0 = __builtin_amdgcn_wmma_f32_16x16x32_bf16(false, aq0, false, bk, (short)0, s0, false, false);
            bk = load_a_bf16(kc + 32, HD_DIM, lane);
            s0 = __builtin_amdgcn_wmma_f32_16x16x32_bf16(false, aq1, false, bk, (short)0, s0, false, false);
            bk = load_a_bf16(kc + 16 * HD_DIM + 0, HD_DIM, lane);
            s1 = __builtin_amdgcn_wmma_f32_16x16x32_bf16(false, aq0, false, bk, (short)0, s1, false, false);
            bk = load_a_bf16(kc + 16 * HD_DIM + 32, HD_DIM, lane);
            s1 = __builtin_amdgcn_wmma_f32_16x16x32_bf16(false, aq1, false, bk, (short)0, s1, false, false);
        }
#pragma unroll
        for (int r = 0; r < 8; ++r) { s0[r] *= scale; s1[r] *= scale; }

        // ---- online softmax: row stats across the 16 lanes holding each row ----
        v8f cm;
#pragma unroll
        for (int r = 0; r < 8; ++r) cm[r] = fmaxf(s0[r], s1[r]);
#pragma unroll
        for (int off = 1; off < 16; off <<= 1) {
#pragma unroll
            for (int r = 0; r < 8; ++r)
                cm[r] = fmaxf(cm[r], __shfl_xor(cm[r], off, 32));
        }
        v8f mn, alpha, p0, p1, cs;
#pragma unroll
        for (int r = 0; r < 8; ++r) {
            mn[r]    = fmaxf(m[r], cm[r]);
            alpha[r] = __expf(m[r] - mn[r]);
            p0[r]    = __expf(s0[r] - mn[r]);
            p1[r]    = __expf(s1[r] - mn[r]);
            cs[r]    = p0[r] + p1[r];
        }
#pragma unroll
        for (int off = 1; off < 16; off <<= 1) {
#pragma unroll
            for (int r = 0; r < 8; ++r)
                cs[r] += __shfl_xor(cs[r], off, 32);
        }
#pragma unroll
        for (int r = 0; r < 8; ++r) { l[r] = l[r] * alpha[r] + cs[r]; m[r] = mn[r]; }
#pragma unroll
        for (int t = 0; t < 4; ++t)
#pragma unroll
            for (int r = 0; r < 8; ++r) acc[t][r] *= alpha[r];

        // ---- transpose P (C layout -> A layout) through per-wave LDS ----
#pragma unroll
        for (int r = 0; r < 8; ++r) {
            tl[(rb + r) * 32 + col]      = (__bf16)p0[r];
            tl[(rb + r) * 32 + 16 + col] = (__bf16)p1[r];
        }
        asm volatile("s_wait_dscnt 0" ::: "memory");
        v16bf ap = load_a_bf16(tl, 32, lane);

        // ---- acc += P * V (V tiles from LDS) ----
#pragma unroll
        for (int t = 0; t < 4; ++t) {
            v16bf bv = load_b_bf16(vc + t * 16, HD_DIM, lane);
            acc[t] = __builtin_amdgcn_wmma_f32_16x16x32_bf16(
                false, ap, false, bv, (short)0, acc[t], false, false);
        }

        buf ^= 1;
    }

    // ---- normalize and store: out[B, S, H*HD] fp32 ----
    const int bb = bh >> 3;
    const int h  = bh & (H_DIM - 1);
    v8f inv;
#pragma unroll
    for (int r = 0; r < 8; ++r) inv[r] = 1.0f / l[r];
#pragma unroll
    for (int t = 0; t < 4; ++t) {
#pragma unroll
        for (int r = 0; r < 8; ++r) {
            const int s = q0 + rb + r;
            const int c = h * HD_DIM + t * 16 + col;
            out[((size_t)bb * S_DIM + s) * D_DIM + c] = acc[t][r] * inv[r];
        }
    }
}

// ---------------------------------------------------------------------------
extern "C" void kernel_launch(void* const* d_in, const int* in_sizes, int n_in,
                              void* d_out, int out_size, void* d_ws, size_t ws_size,
                              hipStream_t stream) {
    (void)in_sizes; (void)n_in; (void)out_size; (void)ws_size;
    const float* q  = (const float*)d_in[0];
    const float* k  = (const float*)d_in[1];
    const float* v  = (const float*)d_in[2];
    const float* wq = (const float*)d_in[3];
    const float* wk = (const float*)d_in[4];
    const float* wv = (const float*)d_in[5];

    __bf16* qkv = (__bf16*)d_ws;  // 3 * B * H * S * HD * 2 bytes = ~12.6 MB

    dim3 g1(B_DIM * S_DIM / 16, D_DIM / 64, 3);   // (256, 8, 3), 1 wave/block
    mha_proj_kernel<<<g1, 32, 0, stream>>>(q, k, v, wq, wk, wv, qkv);

    dim3 g2(B_DIM * H_DIM * (S_DIM / 16) / 8);    // 256 blocks x 8 waves
    mha_attn_kernel<<<g2, 256, 0, stream>>>(qkv, (float*)d_out);
}
